// Temporal_Aggregation_wto_Uncer_18717467476253
// MI455X (gfx1250) — matrix operand order repaired and checked by reading, not compiled, not run on hardware
//
#include <hip/hip_runtime.h>

// MI455X / gfx1250 fused kernel:
//   conv1d(K=3 over T, channels 64->64) -> tridiagonal band-sum over T -> linear + relu
// One wave per (b,n) pair; all GEMMs on v_wmma_f32_16x16x32_f16 with f32 accumulate.
// Weights are pre-swizzled ONCE into d_ws by a 1-block kernel, then async-copied
// (global_load_async_to_lds_b128) into LDS by each block.

typedef __attribute__((ext_vector_type(16))) _Float16 v16h;
typedef __attribute__((ext_vector_type(8)))  _Float16 v8h;
typedef __attribute__((ext_vector_type(4)))  _Float16 v4h;
typedef __attribute__((ext_vector_type(8)))  float    v8f;

#define WAVES      8
#define T_DIM      24
#define D_DIM      64
#define VROWS      34            // padded v rows: index = t+1, t in [-1, 32]
#define YROWS      26            // padded y rows: index = t+1, t in [-1, 24]
#define SB1_HALFS  (24*32*16)    // stage-1 B fragments: 6 ksteps x 4 ntiles
#define SB3_HALFS  (8*32*16)     // stage-3 B fragments: 2 ksteps x 4 ntiles
#define WB_HALFS   (SB1_HALFS + SB3_HALFS)       // 16384 halfs = 32 KB
#define WB_BYTES   (WB_HALFS * 2)
#define SV_HALFS   (VROWS*D_DIM) // per-wave f16 input tile (2176 halfs)
#define SY_FLOATS  (YROWS*D_DIM) // per-wave f32 conv-output tile (1664 floats)
#define SMEM_BYTES ((WB_HALFS + WAVES*SV_HALFS)*2 + WAVES*SY_FLOATS*4)

#define USE_ASYNC_WCOPY 1

__device__ __forceinline__ v16h ld_frag16(const _Float16* p /*16B aligned*/) {
    v8h lo = *(const v8h*)p;
    v8h hi = *(const v8h*)(p + 8);
    v16h r;
#pragma unroll
    for (int e = 0; e < 8; ++e) { r[e] = lo[e]; r[8 + e] = hi[e]; }
    return r;
}

// Compute one pre-swizzled B-fragment lane-row (16 halfs) for stage 1 or stage 3.
// Stage-1 B: lane half hB holds K = 16*hB + e (contiguous); stage-3 B uses AMD table order.
__device__ __forceinline__ void preswizzle_item(int it, const float* __restrict__ Wc,
                                                const float* __restrict__ Wl,
                                                _Float16* __restrict__ dstB1,
                                                _Float16* __restrict__ dstB3) {
    if (it < 768) {
        int f = it >> 5, L = it & 31;
        int s = f >> 2, nt = f & 3;
        int kk = s >> 1, c = s & 1;
        int n = nt * 16 + (L & 15), hB = L >> 4;
        _Float16* dst = dstB1 + (f * 32 + L) * 16;
#pragma unroll
        for (int e = 0; e < 16; ++e) {
            int i = c * 32 + 16 * hB + e;
            dst[e] = (_Float16)Wc[(n * 64 + i) * 3 + kk];
        }
    } else {
        int it2 = it - 768;
        int f = it2 >> 5, L = it2 & 31;
        int s3 = f >> 2, nt = f & 3;
        int n = nt * 16 + (L & 15), hB = L >> 4;
        _Float16* dst = dstB3 + (f * 32 + L) * 16;
#pragma unroll
        for (int e = 0; e < 16; ++e) {
            int d = s3 * 32 + (e < 8 ? 8 * hB + e : 16 + 8 * hB + (e - 8));
            dst[e] = (_Float16)Wl[n * 64 + d];
        }
    }
}

// One-block kernel: gather/swizzle weights into d_ws once per launch.
__global__ void __launch_bounds__(256)
preswizzle_weights(const float* __restrict__ Wc, const float* __restrict__ Wl,
                   _Float16* __restrict__ wsB) {
    for (int it = threadIdx.x; it < (24 + 8) * 32; it += 256)
        preswizzle_item(it, Wc, Wl, wsB, wsB + SB1_HALFS);
}

template <bool PRE>
__global__ void __launch_bounds__(256, 2)
temporal_agg_fused(const float* __restrict__ V,   // (16,1024,24,64)
                   const float* __restrict__ Wc,  // (64,64,1,3)
                   const float* __restrict__ Bc,  // (64,)
                   const float* __restrict__ Wl,  // (64,64)
                   const float* __restrict__ Bl,  // (64,)
                   const _Float16* __restrict__ wsB, // pre-swizzled weights (or null)
                   float* __restrict__ O)         // (16,1024,24,64)
{
    extern __shared__ __align__(16) unsigned char smem[];
    _Float16* sB1  = (_Float16*)smem;                 // [f][lane][16]
    _Float16* sB3  = sB1 + SB1_HALFS;                 // [f][lane][16]
    _Float16* sVa  = sB3 + SB3_HALFS;                 // [wave][VROWS*64]
    float*    sYa  = (float*)(sVa + WAVES*SV_HALFS);  // [wave][YROWS*64]

    const int tid  = threadIdx.x;
    const int wave = tid >> 5;
    const int lane = tid & 31;
    const int m    = lane & 15;   // row (A) / column (B,C,D) within 16-tile
    const int h    = lane >> 4;   // lane half

    // ---- Phase 0: weight fragments into LDS ----
    if (PRE) {
#if USE_ASYNC_WCOPY
        // 32 KB contiguous d_ws -> LDS via CDNA5 async global->LDS (no VGPR landing).
        const unsigned lds_base = (unsigned)(size_t)smem;
        const unsigned long long gsrc = (unsigned long long)(size_t)wsB;
#pragma unroll
        for (int q = 0; q < WB_BYTES / 16 / 256; ++q) {     // 8 chunks/thread
            unsigned byte_off = (unsigned)(tid + q * 256) * 16u;
            unsigned lds_addr = lds_base + byte_off;
            asm volatile("global_load_async_to_lds_b128 %0, %1, %2"
                         :: "v"(lds_addr), "v"(byte_off), "s"(gsrc)
                         : "memory");
        }
#if __has_builtin(__builtin_amdgcn_s_wait_asynccnt)
        __builtin_amdgcn_s_wait_asynccnt(0);
#else
        asm volatile("s_wait_asynccnt 0" ::: "memory");
#endif
#else
        const uint4* src = (const uint4*)wsB;
        uint4* dst = (uint4*)smem;
        for (int q = tid; q < WB_BYTES / 16; q += 256) dst[q] = src[q];
#endif
    } else {
        for (int it = tid; it < (24 + 8) * 32; it += 256)
            preswizzle_item(it, Wc, Wl, sB1, sB3);
    }

    // ---- Phase 1: stage this wave's (b,n) tile into LDS as f16, zero pads ----
    const int pair = blockIdx.x * WAVES + wave;          // 2048 blocks * 8 = 16384
    const float* vg = V + (size_t)pair * (T_DIM * D_DIM);
    _Float16* vt = sVa + wave * SV_HALFS;
    float*    yt = sYa + wave * SY_FLOATS;

    {   // zero v pad rows: row 0 (t=-1) and rows 25..33 (t>=24)
        unsigned int* vt32 = (unsigned int*)vt;          // 2 halfs per word
        vt32[lane] = 0u;                                 // words [0,32)  == row 0
        for (int w = 800 + lane; w < 1088; w += 32) vt32[w] = 0u; // rows 25..33
    }
    for (int i = lane; i < 64; i += 32) {                // y[-1]=0, y[24]=0
        yt[i] = 0.0f;
        yt[25 * 64 + i] = 0.0f;
    }
    {   // 24x64 f32 -> f16, rows 1..24 of vt
        const float4* vg4 = (const float4*)vg;           // 384 float4
#pragma unroll
        for (int q = 0; q < 12; ++q) {
            int idx = lane + 32 * q;
            float4 x = vg4[idx];
            v4h hv = { (_Float16)x.x, (_Float16)x.y, (_Float16)x.z, (_Float16)x.w };
            *(v4h*)(vt + 64 + idx * 4) = hv;
        }
    }
    __syncthreads();

    // ---- Phase 2: stage-1 GEMM  y[t,o] = sum_{kk,i} Wc[o,i,kk] * v[t-1+kk,i] + Bc[o] ----
    // Stage-1 A (this code's order, matching sB1): lane half h holds K = 16h..16h+15 (contiguous).
#pragma unroll
    for (int mt = 0; mt < 2; ++mt) {
        const int Mb = mt * 16;
        v16h A[6];
#pragma unroll
        for (int s = 0; s < 6; ++s) {
            int kk = s >> 1, c = s & 1;
            int row = Mb + m + kk;                       // padded index of t-1+kk
            A[s] = ld_frag16(vt + row * 64 + c * 32 + 16 * h);
        }
#pragma unroll
        for (int nt = 0; nt < 4; ++nt) {
            v8f acc = {0.f, 0.f, 0.f, 0.f, 0.f, 0.f, 0.f, 0.f};
#pragma unroll
            for (int s = 0; s < 6; ++s) {
                v16h b = ld_frag16(sB1 + ((s * 4 + nt) * 32 + lane) * 16);
                acc = __builtin_amdgcn_wmma_f32_16x16x32_f16(
                    false, A[s], false, b, (short)0, acc, false, false);
            }
            const int o = nt * 16 + m;
            const float bc = Bc[o];
#pragma unroll
            for (int r = 0; r < 8; ++r) {
                int t = Mb + 8 * h + r;                  // C/D layout: row = 8h + r
                if (t < T_DIM) yt[(t + 1) * 64 + o] = acc[r] + bc;
            }
        }
    }
    __syncthreads();

    // ---- Phase 3: band-sum u = y[t-1]+y[t]+y[t+1], then linear + relu ----
    float* og = O + (size_t)pair * (T_DIM * D_DIM);
#pragma unroll
    for (int mt = 0; mt < 2; ++mt) {
        const int Mb = mt * 16;
        int t = Mb + m;
        int r0 = t,     r1 = t + 1, r2 = t + 2;          // padded y rows t-1,t,t+1
        if (r0 > 25) r0 = 25;
        if (r1 > 25) r1 = 25;
        if (r2 > 25) r2 = 25;
        v16h A2[2];
#pragma unroll
        for (int s3 = 0; s3 < 2; ++s3) {
            const int d0 = s3 * 32 + 8 * h;
            v16h a;
#pragma unroll
            for (int cb = 0; cb < 2; ++cb) {
                const float* pA = yt + r0 * 64 + d0 + 16 * cb;
                const float* pB = yt + r1 * 64 + d0 + 16 * cb;
                const float* pC = yt + r2 * 64 + d0 + 16 * cb;
#pragma unroll
                for (int j = 0; j < 2; ++j) {
                    float4 a4 = *(const float4*)(pA + 4 * j);
                    float4 b4 = *(const float4*)(pB + 4 * j);
                    float4 c4 = *(const float4*)(pC + 4 * j);
                    int base = cb * 8 + j * 4;
                    a[base + 0] = (_Float16)(a4.x + b4.x + c4.x);
                    a[base + 1] = (_Float16)(a4.y + b4.y + c4.y);
                    a[base + 2] = (_Float16)(a4.z + b4.z + c4.z);
                    a[base + 3] = (_Float16)(a4.w + b4.w + c4.w);
                }
            }
            A2[s3] = a;
        }
#pragma unroll
        for (int nt = 0; nt < 4; ++nt) {
            v8f acc = {0.f, 0.f, 0.f, 0.f, 0.f, 0.f, 0.f, 0.f};
#pragma unroll
            for (int s3 = 0; s3 < 2; ++s3) {
                v16h b = ld_frag16(sB3 + ((s3 * 4 + nt) * 32 + lane) * 16);
                acc = __builtin_amdgcn_wmma_f32_16x16x32_f16(
                    false, A2[s3], false, b, (short)0, acc, false, false);
            }
            const int o = nt * 16 + m;
            const float bl = Bl[o];
#pragma unroll
            for (int r = 0; r < 8; ++r) {
                int to = Mb + 8 * h + r;
                if (to < T_DIM) {
                    float val = acc[r] + bl;
                    og[to * 64 + o] = val > 0.0f ? val : 0.0f;
                }
            }
        }
    }
}

extern "C" void kernel_launch(void* const* d_in, const int* in_sizes, int n_in,
                              void* d_out, int out_size, void* d_ws, size_t ws_size,
                              hipStream_t stream) {
    (void)in_sizes; (void)n_in; (void)out_size;
    const float* V  = (const float*)d_in[0];   // value  (16,1024,24,64)
    const float* Wc = (const float*)d_in[1];   // W_conv (64,64,1,3)
    const float* Bc = (const float*)d_in[2];   // b_conv (64,)
    const float* Wl = (const float*)d_in[3];   // W_lin  (64,64)
    const float* Bl = (const float*)d_in[4];   // b_lin  (64,)
    float* O = (float*)d_out;

    const int pairs  = 16 * 1024;              // B*N
    const int blocks = pairs / WAVES;          // 2048 blocks, 1 wave per (b,n)

    if (ws_size >= (size_t)WB_BYTES) {
        _Float16* wsB = (_Float16*)d_ws;
        hipLaunchKernelGGL(preswizzle_weights, dim3(1), dim3(256), 0, stream,
                           Wc, Wl, wsB);
        hipLaunchKernelGGL((temporal_agg_fused<true>), dim3(blocks), dim3(256),
                           SMEM_BYTES, stream, V, Wc, Bc, Wl, Bl,
                           (const _Float16*)wsB, O);
    } else {
        hipLaunchKernelGGL((temporal_agg_fused<false>), dim3(blocks), dim3(256),
                           SMEM_BYTES, stream, V, Wc, Bc, Wl, Bl,
                           (const _Float16*)nullptr, O);
    }
}
